// HGT_81097572483345
// MI455X (gfx1250) — compile-verified, matrix operand order checked by reading, not run on hardware
//
#include <hip/hip_runtime.h>
#include <math.h>

// ---------------- problem constants ----------------
#define NNEWS 10000
#define NSRC  3000
#define NSRCP 3008         // src rows padded to 16-row multiple (workspace only)
#define FIN   256
#define HCH   128          // HC
#define CO_   64
#define NH    4
#define DH    32
#define NVW   6            // V views
#define NLY   2            // L layers
#define NEDG  80000        // E

typedef __attribute__((ext_vector_type(16))) _Float16 v16h;
typedef __attribute__((ext_vector_type(8)))  float    v8f;

// ---------------- device helpers ----------------
__device__ __forceinline__ float gelu_f(float x) {
  float x3 = x * x * x;
  return 0.5f * x * (1.0f + tanhf(0.7978845608028654f * (x + 0.044715f * x3)));
}
__device__ __forceinline__ float sigm_f(float x) { return 1.0f / (1.0f + expf(-x)); }

// order-preserving float<->uint map for atomicMax-based segment max
__device__ __forceinline__ unsigned fenc(float f) {
  unsigned u = __float_as_uint(f);
  return (u & 0x80000000u) ? ~u : (u | 0x80000000u);
}
__device__ __forceinline__ float fdec(unsigned e) {
  unsigned u = (e & 0x80000000u) ? (e & 0x7FFFFFFFu) : ~e;
  return __uint_as_float(u);
}

// K index pattern for 16-bit WMMA A/B operands (ISA 7.12.2)
__device__ __forceinline__ int kpat(int e, int hi) {
  return ((e & 8) ? 16 : 0) + hi * 8 + (e & 7);
}

// ---------------- generic WMMA GEMM (LDS-staged, templated epilogue) -------
// C[M,N](ldc) = maybe_accum + alpha * act( A[M,K](lda) @ W[K,N](ldw) + bias )
// ACT: 0 none, 1 relu, 2 gelu, 3 elu(x)+1.
// Block = 128 threads = 4 waves sharing one 16-row M tile; wave w owns N
// sub-tile w of a 64-wide macro tile.  OOB rows/cols CLAMPED on load.
// GUARD=false requires C to have ceil(M/16)*16 rows allocated (padded ws).
// Requires: K % 32 == 0, N % 16 == 0, 16B-aligned A/W rows (call sites obey).
template<int ACT, bool BIAS, bool ALPHA, bool ACCUM, bool GUARD>
__global__ void wmma_gemm_k(const float* __restrict__ A, const float* __restrict__ W,
                            const float* __restrict__ bias, float* __restrict__ C,
                            int M, int N, int K, int lda, int ldw, int ldc,
                            const float* __restrict__ alpha_ptr)
{
  __shared__ float sA[16][33];   // 16 rows x 32 k
  __shared__ float sB[32][68];   // 32 k x 64 n (hi/lo halves 32 banks apart)

  int tm   = blockIdx.y;
  int n0   = blockIdx.x * 64;
  int wave = threadIdx.x >> 5;
  int lane = threadIdx.x & 31;
  int mr   = lane & 15;
  int hi   = lane >> 4;

  int sar = threadIdx.x >> 3;           // A stage: row 0..15
  int sac = (threadIdx.x & 7) << 2;     // A stage: col 0..28 step 4
  int arow = tm * 16 + sar; if (arow >= M) arow = M - 1;          // clamp
  const float* Arow = A + (size_t)arow * lda;

  v8f acc = {};
  for (int k0 = 0; k0 < K; k0 += 32) {
    {
      float4 v = *(const float4*)(Arow + k0 + sac);
      sA[sar][sac + 0] = v.x; sA[sar][sac + 1] = v.y;
      sA[sar][sac + 2] = v.z; sA[sar][sac + 3] = v.w;
    }
#pragma unroll
    for (int it = 0; it < 4; ++it) {
      int s  = threadIdx.x + it * 128;  // 0..511 float4 slots
      int kr = s >> 4;                  // k row 0..31
      int c4 = (s & 15) << 2;           // col 0..60 step 4
      int col = n0 + c4; if (col + 3 >= N) col = N - 4;           // clamp (N%4==0)
      float4 v = *(const float4*)(W + (size_t)(k0 + kr) * ldw + col);
      sB[kr][c4 + 0] = v.x; sB[kr][c4 + 1] = v.y;
      sB[kr][c4 + 2] = v.z; sB[kr][c4 + 3] = v.w;
    }
    __syncthreads();
    v16h a, b;
#pragma unroll
    for (int e = 0; e < 16; ++e) {
      int kk = kpat(e, hi);
      a[e] = (_Float16)sA[mr][kk];
      b[e] = (_Float16)sB[kk][wave * 16 + mr];
    }
    acc = __builtin_amdgcn_wmma_f32_16x16x32_f16(false, a, false, b, (short)0, acc,
                                                 false, false);
    __syncthreads();
  }

  if (n0 + wave * 16 >= N) return;      // wave-uniform (N % 16 == 0)
  float alpha = ALPHA ? alpha_ptr[0] : 1.0f;
  int ocol = n0 + wave * 16 + (lane & 15);
  float bv = BIAS ? bias[ocol] : 0.0f;
#pragma unroll
  for (int r = 0; r < 8; ++r) {
    int orow = tm * 16 + r + hi * 8;
    if (!GUARD || orow < M) {
      float v = acc[r] + bv;
      if (ACT == 1)      v = fmaxf(v, 0.0f);
      else if (ACT == 2) v = gelu_f(v);
      else if (ACT == 3) v = (v > 0.0f ? v : (expf(v) - 1.0f)) + 1.0f;
      if (ALPHA) v *= alpha;
      float* p = &C[(size_t)orow * ldc + ocol];
      *p = ACCUM ? (*p + v) : v;
    }
  }
}

// ---------------- elementwise / small kernels ----------------
__global__ void fill_f32(float* p, float v, int n) {
  int i = blockIdx.x * blockDim.x + threadIdx.x;
  if (i < n) p[i] = v;
}
__global__ void gelu_k(const float* __restrict__ in, float* __restrict__ out, int n) {
  int i = blockIdx.x * blockDim.x + threadIdx.x;
  if (i < n) out[i] = gelu_f(in[i]);
}
__global__ void skip_mix_k(float* __restrict__ xd, const float* __restrict__ o,
                           const float* __restrict__ skp, int n) {
  int i = blockIdx.x * blockDim.x + threadIdx.x;
  if (i < n) {
    float sk = sigm_f(skp[0]);
    xd[i] = sk * o[i] + (1.0f - sk) * xd[i];
  }
}
__global__ void softmax_vec_k(const float* in, float* o1, float* o2, int n) {
  if (threadIdx.x != 0 || blockIdx.x != 0) return;
  float m = in[0];
  for (int i = 1; i < n; ++i) m = fmaxf(m, in[i]);
  float e[8], s = 0.0f;
  for (int i = 0; i < n; ++i) { e[i] = expf(in[i] - m); s += e[i]; }
  for (int i = 0; i < n; ++i) { float v = e[i] / s; o1[i] = v; o2[i] = v; }
}

// ---------------- edge (segment softmax + scatter) kernels ----------------
__global__ void edge_alpha_k(const float* __restrict__ qdst, const float* __restrict__ ksrc,
                             const int* __restrict__ si, const int* __restrict__ di,
                             const float* __restrict__ prel, float* __restrict__ alpha, int E)
{
  int idx = blockIdx.x * blockDim.x + threadIdx.x;
  if (idx >= E * NH) return;
  int e = idx / NH, h = idx % NH;
  const float4* q4 = (const float4*)(qdst + (size_t)di[e] * HCH + h * DH);
  const float4* k4 = (const float4*)(ksrc + (size_t)si[e] * HCH + h * DH);
  float acc = 0.0f;
#pragma unroll
  for (int j = 0; j < DH / 4; ++j) {
    float4 a = q4[j], b = k4[j];
    acc += a.x * b.x + a.y * b.y + a.z * b.z + a.w * b.w;
  }
  alpha[idx] = acc * prel[h] * 0.17677669529663687f;   // 1/sqrt(32)
}
__global__ void edge_max_k(const float* __restrict__ alpha, const int* __restrict__ di,
                           unsigned* __restrict__ segmax, int E)
{
  int idx = blockIdx.x * blockDim.x + threadIdx.x;
  if (idx >= E * NH) return;
  int e = idx / NH, h = idx % NH;
  atomicMax(&segmax[(size_t)di[e] * NH + h], fenc(alpha[idx]));
}
__global__ void edge_exp_k(float* __restrict__ alpha, const int* __restrict__ di,
                           const unsigned* __restrict__ segmax, float* __restrict__ segsum, int E)
{
  int idx = blockIdx.x * blockDim.x + threadIdx.x;
  if (idx >= E * NH) return;
  int e = idx / NH, h = idx % NH;
  float m  = fdec(segmax[(size_t)di[e] * NH + h]);
  float ex = expf(alpha[idx] - m);
  alpha[idx] = ex;
  atomicAdd(&segsum[(size_t)di[e] * NH + h], ex);
}
__global__ void edge_msg_k(const float* __restrict__ aexp, const float* __restrict__ segsum,
                           const float* __restrict__ vsrc, const int* __restrict__ si,
                           const int* __restrict__ di, float* __restrict__ outd, int E)
{
  int idx = blockIdx.x * blockDim.x + threadIdx.x;
  if (idx >= E * NH) return;
  int e = idx / NH, h = idx % NH;
  float a = aexp[idx] / (segsum[(size_t)di[e] * NH + h] + 1e-16f);
  const float4* v4 = (const float4*)(vsrc + (size_t)si[e] * HCH + h * DH);
  float* orow = outd + (size_t)di[e] * HCH + h * DH;
#pragma unroll
  for (int j = 0; j < DH / 4; ++j) {
    float4 v = v4[j];
    atomicAdd(&orow[j * 4 + 0], v.x * a);
    atomicAdd(&orow[j * 4 + 1], v.y * a);
    atomicAdd(&orow[j * 4 + 2], v.z * a);
    atomicAdd(&orow[j * 4 + 3], v.w * a);
  }
}

// ---------------- head kernels ----------------
__global__ void th_k(const float* __restrict__ zcat, const float* __restrict__ tw,
                     const float* __restrict__ tb, const float* __restrict__ wt,
                     float* __restrict__ out, int Nr)
{
  int r = blockIdx.x * blockDim.x + threadIdx.x;
  if (r >= Nr) return;
  const float* z = zcat + (size_t)r * (2 * CO_);
  float l0 = tb[0], l1 = tb[1];
  for (int j = 0; j < 2 * CO_; ++j) { l0 += z[j] * tw[j * 2]; l1 += z[j] * tw[j * 2 + 1]; }
  l0 *= wt[0]; l1 *= wt[1];
  float m = fmaxf(l0, l1);
  float e0 = expf(l0 - m), e1 = expf(l1 - m), s = e0 + e1;
  out[(size_t)r * 2]     = e0 / s;
  out[(size_t)r * 2 + 1] = e1 / s;
}
__global__ void std_norm_k(const float* __restrict__ zcat, float* __restrict__ stdv,
                           float* __restrict__ nmean, float* __restrict__ nstd, int Nr)
{
  int r = blockIdx.x * blockDim.x + threadIdx.x;
  if (r >= Nr) return;
  float nm = 0.0f, ns = 0.0f;
  for (int j = 0; j < CO_; ++j) {
    float mv = zcat[(size_t)r * (2 * CO_) + j];
    nm += mv * mv;
    float vv = zcat[(size_t)r * (2 * CO_) + CO_ + j];
    float sv = sqrtf(fmaxf(vv, 1e-24f));
    stdv[(size_t)r * CO_ + j] = sv;
    ns += sv * sv;
  }
  nmean[r] = nm;
  nstd[r]  = ns;
}

// ---------------- cdist + sigmoid (WMMA; LDS A-tile; LDS transpose out) ----
// A_ab[i,j] = sigmoid(||mean_a_i - mean_b_j|| + ||std_a_i - std_b_j||); A_ba = transpose.
// blockIdx.y = i-tile (Na multiple of 16); blockIdx.x covers 4 j-tiles (one per wave).
__global__ void cdist_sig_k(const float* __restrict__ zcat_a, const float* __restrict__ zcat_b,
                            const float* __restrict__ std_a,  const float* __restrict__ std_b,
                            const float* __restrict__ nm_a,   const float* __restrict__ nm_b,
                            const float* __restrict__ ns_a,   const float* __restrict__ ns_b,
                            float* __restrict__ A_ab, float* __restrict__ A_ba,
                            int Na, int Nb)
{
  __shared__ float sM[16][68];                 // mean rows of the i-tile
  __shared__ float sS[16][68];                 // std rows
  __shared__ __align__(16) float sT[4][16][20]; // per-wave 16x16 transpose tile
  int i0 = blockIdx.y * 16;
  for (int s = threadIdx.x; s < 256; s += 128) {      // 256 float4 slots
    int r = s >> 4, c4 = (s & 15) << 2;
    float4 mv = *(const float4*)(zcat_a + (size_t)(i0 + r) * (2 * CO_) + c4);
    sM[r][c4 + 0] = mv.x; sM[r][c4 + 1] = mv.y; sM[r][c4 + 2] = mv.z; sM[r][c4 + 3] = mv.w;
    float4 sv = *(const float4*)(std_a + (size_t)(i0 + r) * CO_ + c4);
    sS[r][c4 + 0] = sv.x; sS[r][c4 + 1] = sv.y; sS[r][c4 + 2] = sv.z; sS[r][c4 + 3] = sv.w;
  }
  __syncthreads();

  int wave = threadIdx.x >> 5;
  int lane = threadIdx.x & 31;
  int jt   = blockIdx.x * 4 + wave;
  int n    = lane & 15;
  int hi   = lane >> 4;
  int j    = jt * 16 + n;
  bool jok = (j < Nb);
  int jc   = jok ? j : (Nb - 1);                       // clamped load row
  const float* bm = zcat_b + (size_t)jc * (2 * CO_);
  const float* bs = std_b  + (size_t)jc * CO_;

  v8f accM = {}, accS = {};
#pragma unroll
  for (int k0 = 0; k0 < CO_; k0 += 32) {
    float4 m0 = *(const float4*)(bm + k0 + hi * 8);
    float4 m1 = *(const float4*)(bm + k0 + hi * 8 + 4);
    float4 m2 = *(const float4*)(bm + k0 + 16 + hi * 8);
    float4 m3 = *(const float4*)(bm + k0 + 16 + hi * 8 + 4);
    float4 s0 = *(const float4*)(bs + k0 + hi * 8);
    float4 s1 = *(const float4*)(bs + k0 + hi * 8 + 4);
    float4 s2 = *(const float4*)(bs + k0 + 16 + hi * 8);
    float4 s3 = *(const float4*)(bs + k0 + 16 + hi * 8 + 4);
    v16h am, av, bmv, bsv;
    bmv[0]=(_Float16)m0.x;  bmv[1]=(_Float16)m0.y;  bmv[2]=(_Float16)m0.z;  bmv[3]=(_Float16)m0.w;
    bmv[4]=(_Float16)m1.x;  bmv[5]=(_Float16)m1.y;  bmv[6]=(_Float16)m1.z;  bmv[7]=(_Float16)m1.w;
    bmv[8]=(_Float16)m2.x;  bmv[9]=(_Float16)m2.y;  bmv[10]=(_Float16)m2.z; bmv[11]=(_Float16)m2.w;
    bmv[12]=(_Float16)m3.x; bmv[13]=(_Float16)m3.y; bmv[14]=(_Float16)m3.z; bmv[15]=(_Float16)m3.w;
    bsv[0]=(_Float16)s0.x;  bsv[1]=(_Float16)s0.y;  bsv[2]=(_Float16)s0.z;  bsv[3]=(_Float16)s0.w;
    bsv[4]=(_Float16)s1.x;  bsv[5]=(_Float16)s1.y;  bsv[6]=(_Float16)s1.z;  bsv[7]=(_Float16)s1.w;
    bsv[8]=(_Float16)s2.x;  bsv[9]=(_Float16)s2.y;  bsv[10]=(_Float16)s2.z; bsv[11]=(_Float16)s2.w;
    bsv[12]=(_Float16)s3.x; bsv[13]=(_Float16)s3.y; bsv[14]=(_Float16)s3.z; bsv[15]=(_Float16)s3.w;
#pragma unroll
    for (int e = 0; e < 16; ++e) {
      int kk = k0 + kpat(e, hi);
      am[e] = (_Float16)sM[n][kk];
      av[e] = (_Float16)sS[n][kk];
    }
    accM = __builtin_amdgcn_wmma_f32_16x16x32_f16(false, am, false, bmv, (short)0, accM,
                                                  false, false);
    accS = __builtin_amdgcn_wmma_f32_16x16x32_f16(false, av, false, bsv, (short)0, accS,
                                                  false, false);
  }

  float vals[8];
#pragma unroll
  for (int r = 0; r < 8; ++r) {
    int i = i0 + r + hi * 8;
    float dm = sqrtf(fmaxf(nm_a[i] + nm_b[jc] - 2.0f * accM[r], 1e-12f));
    float ds = sqrtf(fmaxf(ns_a[i] + ns_b[jc] - 2.0f * accS[r], 1e-12f));
    vals[r] = sigm_f(dm + ds);
  }
  // direct stores: for fixed r, lanes cover 16 consecutive j -> coalesced
  if (jok) {
#pragma unroll
    for (int r = 0; r < 8; ++r)
      A_ab[(size_t)(i0 + r + hi * 8) * Nb + j] = vals[r];
  }
  // transposed stores via LDS: each lane ends up writing two float4 runs
#pragma unroll
  for (int r = 0; r < 8; ++r)
    sT[wave][n][r + hi * 8] = vals[r];
  __syncthreads();
  int jj = lane >> 1;                  // tile row (j offset)
  int hh = lane & 1;                   // which 8-col half
  int jw = jt * 16 + jj;
  float4 t0 = *(const float4*)&sT[wave][jj][hh * 8 + 0];
  float4 t1 = *(const float4*)&sT[wave][jj][hh * 8 + 4];
  if (jw < Nb) {
    float* dst = A_ba + (size_t)jw * Na + i0 + hh * 8;
    *(float4*)(dst + 0) = t0;
    *(float4*)(dst + 4) = t1;
  }
}

// ---------------- host orchestration ----------------
extern "C" void kernel_launch(void* const* d_in, const int* in_sizes, int n_in,
                              void* d_out, int out_size, void* d_ws, size_t ws_size,
                              hipStream_t stream)
{
  (void)out_size; (void)ws_size;
  auto f = [&](int i) { return (const float*)d_in[i]; };

  struct NTP { const float *linw,*linb,*outw,*outb,*meanw,*meanb,*varw,*varb,
                           *lin2w,*lin2b,*tlinw,*tlinb; };
  struct CVP { const float *kw[2],*kb[2],*qw[2],*qb[2],*vw[2],*vb[2],*aw[2],*ab[2],
                           *skp[2],*arel[2],*mrel[2],*prel[2]; };
  const float *x_news = nullptr, *x_src = nullptr, *weight = nullptr, *wtype = nullptr;
  const int *e_ns = nullptr, *e_sn = nullptr;
  NTP nt[2]; CVP cv[NLY];

  bool sorted_scheme = (n_in >= 1 && in_sizes[0] == 2 * NEDG);
  if (sorted_scheme) {                 // jax sorted-key leaf order
    e_ns = (const int*)d_in[0]; e_sn = (const int*)d_in[1];
    int idx = 2;
    for (int l = 0; l < NLY; ++l) {
      CVP& c = cv[l];
      c.ab[0]=f(idx++);   c.ab[1]=f(idx++);
      c.arel[0]=f(idx++); c.arel[1]=f(idx++);
      c.aw[0]=f(idx++);   c.aw[1]=f(idx++);
      c.kb[0]=f(idx++);   c.kb[1]=f(idx++);
      c.kw[0]=f(idx++);   c.kw[1]=f(idx++);
      c.mrel[0]=f(idx++); c.mrel[1]=f(idx++);
      c.prel[0]=f(idx++); c.prel[1]=f(idx++);
      c.qb[0]=f(idx++);   c.qb[1]=f(idx++);
      c.qw[0]=f(idx++);   c.qw[1]=f(idx++);
      c.skp[0]=f(idx++);  c.skp[1]=f(idx++);
      c.vb[0]=f(idx++);   c.vb[1]=f(idx++);
      c.vw[0]=f(idx++);   c.vw[1]=f(idx++);
    }
    for (int t = 0; t < 2; ++t) {
      NTP& p = nt[t];
      p.lin2b=f(idx++); p.lin2w=f(idx++); p.linb=f(idx++);  p.linw=f(idx++);
      p.meanb=f(idx++); p.meanw=f(idx++); p.outb=f(idx++);  p.outw=f(idx++);
      p.tlinb=f(idx++); p.tlinw=f(idx++); p.varb=f(idx++);  p.varw=f(idx++);
    }
    weight=f(idx++); wtype=f(idx++); x_news=f(idx++); x_src=f(idx++);
  } else {                             // dict insertion order
    x_news = f(0); x_src = f(1);
    e_ns = (const int*)d_in[2]; e_sn = (const int*)d_in[3];
    weight = f(4); wtype = f(5);
    int idx = 6;
    for (int t = 0; t < 2; ++t) {
      NTP& p = nt[t];
      p.linw=f(idx++);  p.linb=f(idx++);  p.outw=f(idx++);  p.outb=f(idx++);
      p.meanw=f(idx++); p.meanb=f(idx++); p.varw=f(idx++);  p.varb=f(idx++);
      p.lin2w=f(idx++); p.lin2b=f(idx++); p.tlinw=f(idx++); p.tlinb=f(idx++);
    }
    for (int l = 0; l < NLY; ++l) {
      CVP& c = cv[l];
      c.kw[0]=f(idx++);   c.kw[1]=f(idx++);   c.kb[0]=f(idx++);   c.kb[1]=f(idx++);
      c.qw[0]=f(idx++);   c.qw[1]=f(idx++);   c.qb[0]=f(idx++);   c.qb[1]=f(idx++);
      c.vw[0]=f(idx++);   c.vw[1]=f(idx++);   c.vb[0]=f(idx++);   c.vb[1]=f(idx++);
      c.aw[0]=f(idx++);   c.aw[1]=f(idx++);   c.ab[0]=f(idx++);   c.ab[1]=f(idx++);
      c.skp[0]=f(idx++);  c.skp[1]=f(idx++);
      c.arel[0]=f(idx++); c.arel[1]=f(idx++);
      c.mrel[0]=f(idx++); c.mrel[1]=f(idx++);
      c.prel[0]=f(idx++); c.prel[1]=f(idx++);
    }
  }

  // ---- workspace carve-out (floats); src buffers padded to NSRCP rows ----
  const int NN[2]  = { NNEWS, NSRC  };    // real rows
  const int NNA[2] = { NNEWS, NSRCP };    // allocated rows
  float* ws = (float*)d_ws;
  size_t off = 0;
  auto alloc = [&](size_t nel) { float* p = ws + off; off += nel; return p; };
  float* xd[2]    = { alloc((size_t)NNA[0]*HCH), alloc((size_t)NNA[1]*HCH) };
  float* kbuf[2]  = { alloc((size_t)NNA[0]*HCH), alloc((size_t)NNA[1]*HCH) };
  float* qbuf[2]  = { alloc((size_t)NNA[0]*HCH), alloc((size_t)NNA[1]*HCH) };
  float* vbuf[2]  = { alloc((size_t)NNA[0]*HCH), alloc((size_t)NNA[1]*HCH) };
  float* oacc[2]  = { alloc((size_t)NNA[0]*HCH), alloc((size_t)NNA[1]*HCH) };
  float* ksrcT[2] = { alloc((size_t)NNA[0]*HCH), alloc((size_t)NNA[1]*HCH) };
  float* vsrcT[2] = { alloc((size_t)NNA[0]*HCH), alloc((size_t)NNA[1]*HCH) };
  float* gbuf[2]  = { ksrcT[0], ksrcT[1] };   // reuse after message passes
  float* obuf[2]  = { vsrcT[0], vsrcT[1] };
  float* alphaB   = alloc((size_t)NEDG*NH);
  float* segmaxB  = alloc((size_t)NNEWS*NH);
  float* segsumB  = alloc((size_t)NNEWS*NH);
  float* Z[2]     = { alloc((size_t)NNA[0]*CO_),   alloc((size_t)NNA[1]*CO_) };
  float* zcat[2]  = { alloc((size_t)NNA[0]*2*CO_), alloc((size_t)NNA[1]*2*CO_) };
  float* stdv[2]  = { alloc((size_t)NNA[0]*CO_),   alloc((size_t)NNA[1]*CO_) };
  float* nmean[2] = { alloc(NNA[0]), alloc(NNA[1]) };
  float* nstd[2]  = { alloc(NNA[0]), alloc(NNA[1]) };
  float* wnB      = alloc(8);
  float* wtB      = alloc(8);

  // ---- output carve-out ----
  float* out  = (float*)d_out;
  float* A_ns = out;
  float* A_sn = A_ns + (size_t)NNEWS*NSRC;
  float* Xh[2]; Xh[0] = A_sn + (size_t)NNEWS*NSRC; Xh[1] = Xh[0] + (size_t)NNEWS*FIN;
  float* Th[2]; Th[0] = Xh[1] + (size_t)NSRC*FIN;  Th[1] = Th[0] + (size_t)NNEWS*2;
  float* wnO = Th[1] + (size_t)NSRC*2;
  float* wtO = wnO + NVW;

  // dispatch: only the template combinations actually used
  auto gemm = [&](const float* A, const float* W, const float* b, float* C,
                  int M, int N, int K, int lda, int ldw, int ldc,
                  const float* al, int act, int acc, bool guard) {
    dim3 grid((N + 63) / 64, (M + 15) / 16);
#define GL(ACT,BIAS,ALPHA,ACCUM,GUARD) \
    wmma_gemm_k<ACT,BIAS,ALPHA,ACCUM,GUARD><<<grid,128,0,stream>>>( \
        A,W,b,C,M,N,K,lda,ldw,ldc,al)
    if (act == 1)          GL(1, true,  false, false, false);
    else if (act == 3)     GL(3, true,  false, false, false);
    else if (al && acc)    GL(0, true,  true,  true,  false);
    else if (al)           GL(0, true,  true,  false, false);
    else if (!b)           GL(0, false, false, false, false);
    else if (guard)        GL(0, true,  false, false, true );
    else                   GL(0, true,  false, false, false);
#undef GL
  };
  auto fill = [&](float* p, float v, int n) {
    fill_f32<<<(n + 255) / 256, 256, 0, stream>>>(p, v, n);
  };

  softmax_vec_k<<<1, 1, 0, stream>>>(weight, wnB, wnO, NVW);
  softmax_vec_k<<<1, 1, 0, stream>>>(wtype,  wtB, wtO, 2);

  const int eb = (NEDG * NH + 255) / 256;
  for (int i = 0; i < NVW; ++i) {
    gemm(x_news + (size_t)i*NNEWS*FIN, nt[0].linw + (size_t)i*FIN*HCH, nt[0].linb + i*HCH,
         xd[0], NNEWS, HCH, FIN, FIN, HCH, HCH, nullptr, 1, 0, false);
    gemm(x_src  + (size_t)i*NSRC*FIN,  nt[1].linw + (size_t)i*FIN*HCH, nt[1].linb + i*HCH,
         xd[1], NSRC, HCH, FIN, FIN, HCH, HCH, nullptr, 1, 0, false);

    for (int l = 0; l < NLY; ++l) {
      CVP& c = cv[l];
      for (int t = 0; t < 2; ++t) {
        gemm(xd[t], c.kw[t], c.kb[t], kbuf[t], NN[t], HCH, HCH, HCH, HCH, HCH,
             nullptr, 0, 0, false);
        gemm(xd[t], c.qw[t], c.qb[t], qbuf[t], NN[t], HCH, HCH, HCH, HCH, HCH,
             nullptr, 0, 0, false);
        gemm(xd[t], c.vw[t], c.vb[t], vbuf[t], NN[t], HCH, HCH, HCH, HCH, HCH,
             nullptr, 0, 0, false);
        fill(oacc[t], 0.0f, NN[t] * HCH);
      }
      for (int et = 0; et < 2; ++et) {     // et0: news->src ('ns'); et1: src->news ('sn')
        int ts = (et == 0) ? 0 : 1;
        int td = 1 - ts;
        const int* si = (et == 0) ? e_ns : e_sn;
        const int* di = si + NEDG;
        for (int h = 0; h < NH; ++h) {     // per-head 32x32 rel transforms via WMMA
          gemm(kbuf[ts] + h*DH, c.arel[et] + (size_t)h*DH*DH, nullptr, ksrcT[et] + h*DH,
               NN[ts], DH, DH, HCH, DH, HCH, nullptr, 0, 0, false);
          gemm(vbuf[ts] + h*DH, c.mrel[et] + (size_t)h*DH*DH, nullptr, vsrcT[et] + h*DH,
               NN[ts], DH, DH, HCH, DH, HCH, nullptr, 0, 0, false);
        }
        fill(segmaxB, 0.0f, NN[td] * NH);  // uint 0 == below -inf under encoding
        fill(segsumB, 0.0f, NN[td] * NH);
        edge_alpha_k<<<eb, 256, 0, stream>>>(qbuf[td], ksrcT[et], si, di, c.prel[et],
                                             alphaB, NEDG);
        edge_max_k<<<eb, 256, 0, stream>>>(alphaB, di, (unsigned*)segmaxB, NEDG);
        edge_exp_k<<<eb, 256, 0, stream>>>(alphaB, di, (const unsigned*)segmaxB,
                                           segsumB, NEDG);
        edge_msg_k<<<eb, 256, 0, stream>>>(alphaB, segsumB, vsrcT[et], si, di,
                                           oacc[td], NEDG);
      }
      for (int t = 0; t < 2; ++t) {
        int n = NN[t] * HCH;
        gelu_k<<<(n + 255) / 256, 256, 0, stream>>>(oacc[t], gbuf[t], n);
        gemm(gbuf[t], c.aw[t], c.ab[t], obuf[t], NN[t], HCH, HCH, HCH, HCH, HCH,
             nullptr, 0, 0, false);
        skip_mix_k<<<(n + 255) / 256, 256, 0, stream>>>(xd[t], obuf[t], c.skp[t], n);
      }
    }
    for (int t = 0; t < 2; ++t)
      gemm(xd[t], nt[t].outw, nt[t].outb, Z[t], NN[t], CO_, HCH, HCH, CO_, CO_,
           wnB + i, 0, (i > 0) ? 1 : 0, false);
  }

  for (int t = 0; t < 2; ++t) {
    gemm(Z[t], nt[t].meanw, nt[t].meanb, zcat[t],       NN[t], CO_, CO_, CO_, CO_, 2*CO_,
         nullptr, 0, 0, false);
    gemm(Z[t], nt[t].varw,  nt[t].varb,  zcat[t] + CO_, NN[t], CO_, CO_, CO_, CO_, 2*CO_,
         nullptr, 3, 0, false);
    gemm(zcat[t], nt[t].lin2w, nt[t].lin2b, Xh[t], NN[t], FIN, 2*CO_, 2*CO_, FIN, FIN,
         nullptr, 0, 0, t == 1);   // Xh_src lives in d_out: guard partial M tile
    th_k<<<(NN[t] + 255) / 256, 256, 0, stream>>>(zcat[t], nt[t].tlinw, nt[t].tlinb,
                                                  wtB, Th[t], NN[t]);
    std_norm_k<<<(NN[t] + 255) / 256, 256, 0, stream>>>(zcat[t], stdv[t], nmean[t],
                                                        nstd[t], NN[t]);
  }

  dim3 cgrid((((NSRC + 15) / 16) + 3) / 4, NNEWS / 16);
  cdist_sig_k<<<cgrid, 128, 0, stream>>>(zcat[0], zcat[1], stdv[0], stdv[1],
                                         nmean[0], nmean[1], nstd[0], nstd[1],
                                         A_ns, A_sn, NNEWS, NSRC);
}
